// ActorCriticNet_45561013076593
// MI455X (gfx1250) — matrix operand order repaired and checked by reading, not compiled, not run on hardware
//
#include <hip/hip_runtime.h>
#include <hip/hip_bf16.h>

// ---------------------------------------------------------------------------
// GCN actor-critic forward for MI455X (gfx1250, wave32).
//   3x [ scatter-add over edges -> f32->bf16 pass -> relu(agg @ W + b) WMMA ]
//   then mean-over-nodes + 3 small heads.
// ---------------------------------------------------------------------------

#define N_NODES 50000
#define HID     128

typedef __attribute__((ext_vector_type(16))) __bf16          v16bf;
typedef __attribute__((ext_vector_type(16))) unsigned short  v16us;
typedef __attribute__((ext_vector_type(8)))  unsigned short  v8us;
typedef __attribute__((ext_vector_type(8)))  float           v8f;

__device__ __forceinline__ unsigned short f32_to_bf16_rne(float f) {
    unsigned int u = __builtin_bit_cast(unsigned int, f);
    unsigned int r = u + 0x7FFFu + ((u >> 16) & 1u);   // round-to-nearest-even
    return (unsigned short)(r >> 16);
}

// ---------------------------------------------------------------------------
// Edge scatter-add: one wave per edge; 32 lanes x float4 = 128 dims.
// h (25.6MB) and agg (25.6MB) both live in the 192MB L2, so the gathers and
// the atomics resolve at L2 bandwidth; HBM only streams the edge indices.
// ---------------------------------------------------------------------------
__global__ void scatter_edges_kernel(const float* __restrict__ H,
                                     const int*   __restrict__ src,
                                     const int*   __restrict__ dst,
                                     float*       __restrict__ Agg,
                                     int n_edges) {
    int e = blockIdx.x * 8 + (threadIdx.x >> 5);
    if (e >= n_edges) return;
    int lane = threadIdx.x & 31;
    int s = src[e];
    int d = dst[e];
    const float4 v = *(const float4*)(H + (size_t)s * HID + lane * 4);
    float* p = Agg + (size_t)d * HID + lane * 4;
    unsafeAtomicAdd(p + 0, v.x);   // global_atomic_add_f32 (no CAS loop)
    unsafeAtomicAdd(p + 1, v.y);
    unsafeAtomicAdd(p + 2, v.z);
    unsafeAtomicAdd(p + 3, v.w);
}

// ---------------------------------------------------------------------------
// One-pass f32 -> bf16 conversion of the aggregated features. Keeps the
// conversion VALU work out of the WMMA kernel (where it was done 8x
// redundantly per element) and halves the GEMM's A-side load bytes.
// ---------------------------------------------------------------------------
__global__ void convert_a_kernel(const float* __restrict__ A,
                                 unsigned short* __restrict__ Ab, int n4) {
    int i = blockIdx.x * 256 + threadIdx.x;       // one float4 per thread
    if (i >= n4) return;
    float4 f = ((const float4*)A)[i];
    uint2 o;
    o.x = (unsigned int)f32_to_bf16_rne(f.x) | ((unsigned int)f32_to_bf16_rne(f.y) << 16);
    o.y = (unsigned int)f32_to_bf16_rne(f.z) | ((unsigned int)f32_to_bf16_rne(f.w) << 16);
    ((uint2*)Ab)[i] = o;
}

// ---------------------------------------------------------------------------
// Convert W (f32, [k][n] row-major) -> Wt (bf16 bits, [n][k] row-major).
// Transposed layout makes the WMMA B-fragment load contiguous per lane.
// ---------------------------------------------------------------------------
__global__ void convert_wt_kernel(const float* __restrict__ W,
                                  unsigned short* __restrict__ Wt) {
    int idx = blockIdx.x * 256 + threadIdx.x;     // 16384 elements
    int k = idx >> 7;
    int n = idx & 127;
    Wt[n * HID + k] = f32_to_bf16_rne(W[k * HID + n]);
}

// ---------------------------------------------------------------------------
// Out = relu(A @ W + bias), A: bf16 [50000,128] row-major, Wt: bf16 [n][k].
// Grid: 3125 blocks (one 16-row M tile), 256 threads = 8 waves; wave w
// computes C tile (mtile, ntile=w). K loop: 4 x v_wmma_f32_16x16x32_bf16.
//
// Fragment layouts used here (wave32; A and B share the contiguous k order):
//   A 16x32 bf16 : lane l -> row m=l&15; halves[0..15]=K kbase+hi*16+(0..15), hi=l>>4.
//   B 32x16 bf16 : lane l -> col n=l&15; halves[0..15]=K kbase+hi*16+(0..15).
//   C/D 16x16 f32: lane l -> col n=l&15; vgpr r -> row m=(l>>4)*8+r.
// ---------------------------------------------------------------------------
__global__ void gcn_gemm_relu_kernel(const unsigned short* __restrict__ Ab,
                                     const unsigned short* __restrict__ Wt,
                                     const float* __restrict__ bias,
                                     float* __restrict__ Out) {
    const int mtile = blockIdx.x;           // 0..3124 (50000 = 3125*16 exact)
    const int ntile = threadIdx.x >> 5;     // 0..7 (wave id)
    const int lane  = threadIdx.x & 31;
    const int lmod  = lane & 15;
    const int lhi   = lane >> 4;            // 0/1

    const unsigned short* ap = Ab + (size_t)(mtile * 16 + lmod) * HID;
    const unsigned short* bp = Wt + (size_t)(ntile * 16 + lmod) * HID;

    v8f c = {};
#pragma unroll
    for (int kk = 0; kk < 4; ++kk) {
        const int kbase = kk * 32;

        // A fragment: 16 contiguous bf16 (32B, 32B-aligned), K kbase+lhi*16+(0..15).
        v16us au = *(const v16us*)(ap + kbase + lhi * 16);

        // B fragment: 16 contiguous bf16 (32B, 32B-aligned).
        v16us bu = *(const v16us*)(bp + kbase + lhi * 16);

        c = __builtin_amdgcn_wmma_f32_16x16x32_bf16(
                /*neg_a=*/false, __builtin_bit_cast(v16bf, au),
                /*neg_b=*/false, __builtin_bit_cast(v16bf, bu),
                /*c_mod=*/(short)0, c,
                /*reuse_a=*/false, /*reuse_b=*/false);
    }

    const float bn = bias[ntile * 16 + lmod];
    float* outp = Out + (size_t)(mtile * 16) * HID + ntile * 16 + lmod;
#pragma unroll
    for (int r = 0; r < 8; ++r) {
        int m = lhi * 8 + r;
        float v = c[r] + bn;
        outp[(size_t)m * HID] = v > 0.0f ? v : 0.0f;
    }
}

// ---------------------------------------------------------------------------
// Column sums for mean_nodes: each block reduces a 128-row slab, one atomic
// per column per block (50k atomics total -> negligible).
// ---------------------------------------------------------------------------
__global__ void col_sum_kernel(const float* __restrict__ H,
                               float* __restrict__ msum, int n_rows) {
    __shared__ float sh[HID];
    int col  = threadIdx.x & 127;
    int half = threadIdx.x >> 7;                // 0/1
    int end  = blockIdx.x * 128 + 128;
    if (end > n_rows) end = n_rows;
    float acc = 0.0f;
    for (int r = blockIdx.x * 128 + half; r < end; r += 2)
        acc += H[(size_t)r * HID + col];
    if (half) sh[col] = acc;
    __syncthreads();
    if (!half) unsafeAtomicAdd(&msum[col], acc + sh[col]);
}

// ---------------------------------------------------------------------------
// Finalize heads: mN = msum/N; v = mN.w_v + b_v; pi_done = mN.w_pd + b_pd.
// Output layout: [0..N-1]=pi_graph, [N]=pi_done, [N+1]=v.
// ---------------------------------------------------------------------------
__global__ void finalize_kernel(const float* __restrict__ msum,
                                const float* __restrict__ w_v,  const float* __restrict__ b_v,
                                const float* __restrict__ w_pd, const float* __restrict__ b_pd,
                                float* __restrict__ out) {
    __shared__ float sv[HID];
    __shared__ float sd[HID];
    int t = threadIdx.x;                       // 128 threads
    float m = msum[t] * (1.0f / (float)N_NODES);
    sv[t] = m * w_v[t];
    sd[t] = m * w_pd[t];
    __syncthreads();
    for (int s = 64; s > 0; s >>= 1) {
        if (t < s) { sv[t] += sv[t + s]; sd[t] += sd[t + s]; }
        __syncthreads();
    }
    if (t == 0) {
        out[N_NODES]     = sd[0] + b_pd[0];    // pi_done
        out[N_NODES + 1] = sv[0] + b_v[0];     // v
    }
}

// ---------------------------------------------------------------------------
// pi_graph = h @ w_pg + b_pg : one thread per node, float4 dot of 128.
// ---------------------------------------------------------------------------
__global__ void pi_graph_kernel(const float* __restrict__ H,
                                const float* __restrict__ w,
                                const float* __restrict__ b,
                                float* __restrict__ out, int n) {
    int i = blockIdx.x * 256 + threadIdx.x;
    if (i >= n) return;
    const float4* hp = (const float4*)(H + (size_t)i * HID);
    const float4* wp = (const float4*)w;
    float acc = 0.0f;
#pragma unroll
    for (int j = 0; j < HID / 4; ++j) {
        float4 hv = hp[j], wv = wp[j];
        acc += hv.x * wv.x + hv.y * wv.y + hv.z * wv.z + hv.w * wv.w;
    }
    out[i] = acc + b[0];
}

// ---------------------------------------------------------------------------
extern "C" void kernel_launch(void* const* d_in, const int* in_sizes, int n_in,
                              void* d_out, int out_size, void* d_ws, size_t ws_size,
                              hipStream_t stream) {
    const float* x    = (const float*)d_in[0];
    const int*   src  = (const int*)  d_in[1];
    const int*   dst  = (const int*)  d_in[2];
    const float* W[3] = { (const float*)d_in[3], (const float*)d_in[5], (const float*)d_in[7] };
    const float* B[3] = { (const float*)d_in[4], (const float*)d_in[6], (const float*)d_in[8] };
    const float* w_pg = (const float*)d_in[9];
    const float* b_pg = (const float*)d_in[10];
    const float* w_pd = (const float*)d_in[11];
    const float* b_pd = (const float*)d_in[12];
    const float* w_v  = (const float*)d_in[13];
    const float* b_v  = (const float*)d_in[14];
    float* out = (float*)d_out;

    const int n_edges = in_sizes[1];
    const int n_feat  = N_NODES * HID;                       // 6.4M
    const size_t feat_bytes = (size_t)n_feat * sizeof(float); // 25.6 MB

    char* ws = (char*)d_ws;
    float*          agg  = (float*)ws;                                   // 25.6 MB
    float*          h    = (float*)(ws + feat_bytes);                    // 25.6 MB
    unsigned short* Abf  = (unsigned short*)(ws + 2 * feat_bytes);       // 12.8 MB
    unsigned short* Wt   = (unsigned short*)(ws + 2 * feat_bytes + feat_bytes / 2);
    float*          msum = (float*)(ws + 2 * feat_bytes + feat_bytes / 2
                                    + (size_t)HID * HID * sizeof(unsigned short));

    const float* hin = x;
    for (int layer = 0; layer < 3; ++layer) {
        hipMemsetAsync(agg, 0, feat_bytes, stream);
        scatter_edges_kernel<<<(n_edges + 7) / 8, 256, 0, stream>>>(hin, src, dst, agg, n_edges);
        convert_a_kernel<<<(n_feat / 4 + 255) / 256, 256, 0, stream>>>(agg, Abf, n_feat / 4);
        convert_wt_kernel<<<(HID * HID) / 256, 256, 0, stream>>>(W[layer], Wt);
        gcn_gemm_relu_kernel<<<N_NODES / 16, 256, 0, stream>>>(Abf, Wt, B[layer], h);
        hin = h;
    }

    hipMemsetAsync(msum, 0, HID * sizeof(float), stream);
    col_sum_kernel<<<(N_NODES + 127) / 128, 256, 0, stream>>>(h, msum, N_NODES);
    finalize_kernel<<<1, HID, 0, stream>>>(msum, w_v, b_v, w_pd, b_pd, out);
    pi_graph_kernel<<<(N_NODES + 255) / 256, 256, 0, stream>>>(h, w_pg, b_pg, out, N_NODES);
}
